// StudentMamba2_39281770889621
// MI455X (gfx1250) — compile-verified, matrix operand order checked by reading, not run on hardware
//
#include <hip/hip_runtime.h>
#include <math.h>
#include <stdint.h>

// ---------------------------------------------------------------------------
// CDNA5 (gfx1250) hybrid Mamba/Attn/MoE forward.
//  - All GEMMs: v_wmma_f32_16x16x32_bf16 (wave32), f32 accumulate.
//  - Double-buffered LDS pipeline:
//      A tiles  : per-lane GLOBAL_LOAD_ASYNC_TO_LDS_B128   (ASYNCcnt)
//      B tiles  : TENSOR_LOAD_TO_LDS (TDM D# with LDS pad = 80B row stride,
//                 issued by wave 0)                         (TENSORcnt)
//    next tile's DMA overlaps current tile's WMMAs; drains use in-order
//    completion semantics (ISA ch.10.7 / ch.7 TDM).
// ---------------------------------------------------------------------------

typedef __bf16 bf16_t;
typedef __attribute__((ext_vector_type(16))) bf16_t        v16bf;
typedef __attribute__((ext_vector_type(8)))  float         v8f;
typedef __attribute__((ext_vector_type(8)))  unsigned int  v8u;
typedef __attribute__((ext_vector_type(4)))  unsigned int  v4u;
typedef __attribute__((ext_vector_type(8)))  int           v8i;
typedef __attribute__((ext_vector_type(4)))  int           v4i;

#define BM   128
#define BN   64
#define BK   32
#define LDSS 40   // LDS row stride in 16-bit units (80 B, 16B aligned)

#if __has_builtin(__builtin_amdgcn_tensor_load_to_lds)
#define HYB_TDM 1
#else
#define HYB_TDM 0
#endif

__device__ __forceinline__ unsigned short f2bf(float f) {
  union { float f; unsigned int u; } c; c.f = f;
  unsigned int u = c.u;
  if ((u & 0x7F800000u) == 0x7F800000u) return (unsigned short)(u >> 16); // inf/nan
  return (unsigned short)((u + 0x7FFFu + ((u >> 16) & 1u)) >> 16);        // RNE
}

// Build one wave32 WMMA fragment (16 bf16/lane) from an LDS row.
// Pair order per ISA 7.12.2 (16-bit A 16x32): K{kb+0..7}, then K{kb+16..23}.
__device__ __forceinline__ v16bf frag_from_lds(const unsigned short* base) {
  v8u u;
  u[0] = *(const unsigned int*)(base + 0);
  u[1] = *(const unsigned int*)(base + 2);
  u[2] = *(const unsigned int*)(base + 4);
  u[3] = *(const unsigned int*)(base + 6);
  u[4] = *(const unsigned int*)(base + 16);
  u[5] = *(const unsigned int*)(base + 18);
  u[6] = *(const unsigned int*)(base + 20);
  u[7] = *(const unsigned int*)(base + 22);
  return __builtin_bit_cast(v16bf, u);
}

// C = act(A[M,K](bf16) @ B + bias); B bf16 [K,N] (transB=0) or [N,K] (transB=1).
// Outputs: C (f32, optional) and/or Cbf (bf16, optional). act: 0=none, 1=silu.
__global__ __launch_bounds__(256) void hyb_gemm_bf16(
    const unsigned short* __restrict__ A, int lda,
    const unsigned short* __restrict__ B, int ldb, int transB,
    const float* __restrict__ bias,
    float* __restrict__ C, int ldc,
    unsigned short* __restrict__ Cbf, int ldcb,
    int M, int N, int K, int act) {
  __shared__ __align__(16) unsigned short As[2][BM * LDSS];  // [m][k]
  __shared__ __align__(16) unsigned short Bs[2][BN * LDSS];  // [n][k]

  const int tid  = threadIdx.x;
  const int lane = tid & 31;
  const int wid  = tid >> 5;
  const int rowBase = blockIdx.y * BM;
  const int colBase = blockIdx.x * BN;

  const bool aRowsIn = (rowBase + BM <= M);   // uniform
  const bool bColsIn = (colBase + BN <= N);   // uniform
  const int  ntiles  = (K + BK - 1) / BK;

  // ---- stage A tile; returns #async instructions issued per wave (0 or 2) ----
  auto stageA = [&](int k0, int buf) -> int {
    if (aRowsIn && (k0 + BK <= K)) {
#pragma unroll
      for (int si = 0; si < 2; ++si) {
        int s  = tid * 2 + si;
        int r  = s >> 2;
        int cs = (s & 3) * 8;
        unsigned ldsa = (unsigned)(size_t)&As[buf][r * LDSS + cs];
        const unsigned short* g = A + (size_t)(rowBase + r) * lda + k0 + cs;
        asm volatile("global_load_async_to_lds_b128 %0, %1, off"
                     :: "v"(ldsa), "v"(g) : "memory");
      }
      return 2;
    }
    // edge: clamped (always-valid) loads + branch-free zero select
#pragma unroll
    for (int si = 0; si < 2; ++si) {
      int s  = tid * 2 + si;
      int r  = s >> 2;
      int cs = (s & 3) * 8;
      int gr = rowBase + r;
      int grc = gr < M ? gr : M - 1;
      unsigned short* dst = &As[buf][r * LDSS + cs];
#pragma unroll
      for (int j = 0; j < 8; ++j) {
        int gk  = k0 + cs + j;
        int gkc = gk < K ? gk : K - 1;
        unsigned short v = A[(size_t)grc * lda + gkc];
        dst[j] = (gr < M && gk < K) ? v : (unsigned short)0;
      }
    }
    return 0;
  };

  // ---- stage B tile; returns 1 if a TDM tensor-load was issued (wave 0) ----
  auto stageB = [&](int k0, int buf) -> int {
    if (transB) {
      if (bColsIn && (k0 + BK <= K)) {
#if HYB_TDM
        if (wid == 0) {
          unsigned long long ga =
              (unsigned long long)(B + (size_t)colBase * ldb + k0);
          unsigned ldsb = (unsigned)(size_t)&Bs[buf][0];
          // D# group0: count=1, lds_addr, global_addr[56:0], type=2
          v4u g0 = { 1u, ldsb, (unsigned)(ga & 0xFFFFFFFFull),
                     (unsigned)((ga >> 32) & 0x01FFFFFFull) | (2u << 30) };
          // D# group1: data_size=2B(code 1), pad_enable,
          //   pad_interval=3 (64B), pad_amount=3 (16B) -> 80B LDS row stride;
          //   tensor_dim0/1 = large; tile 32(k) x 64(n); dim0_stride=ldb.
          v8i g1 = { (int)((1u << 16) | (1u << 20) | (3u << 22) | (3u << 25)),
                     (int)(0xFFFFu << 16),              // tensor_dim0 lo16
                     (int)(0x7FFFu | (0xFFFFu << 16)),  // dim0 hi16, dim1 lo16
                     (int)(0x7FFFu | (32u << 16)),      // dim1 hi16, tile_dim0
                     (int)64u,                          // tile_dim1 (tile_dim2=0)
                     (int)ldb,                          // dim0_stride[31:0]
                     0, 0 };
          v4i g2 = { 0, 0, 0, 0 };
          v4i g3 = { 0, 0, 0, 0 };
          v8i g4 = { 0, 0, 0, 0, 0, 0, 0, 0 };
          __builtin_amdgcn_tensor_load_to_lds(g0, g1, g2, g3, g4, 0);
        }
        return 1;
#else
        int nr = tid >> 2;
        int ks = (tid & 3) * 8;
        unsigned ldsb = (unsigned)(size_t)&Bs[buf][nr * LDSS + ks];
        const unsigned short* g = B + (size_t)(colBase + nr) * ldb + k0 + ks;
        asm volatile("global_load_async_to_lds_b128 %0, %1, off"
                     :: "v"(ldsb), "v"(g) : "memory");
        return 0;
#endif
      }
      // edge: clamped + zero-select
      int nr = tid >> 2;
      int ks = (tid & 3) * 8;
      int gn = colBase + nr;
      int gnc = gn < N ? gn : N - 1;
      unsigned short* dst = &Bs[buf][nr * LDSS + ks];
#pragma unroll
      for (int j = 0; j < 8; ++j) {
        int gk  = k0 + ks + j;
        int gkc = gk < K ? gk : K - 1;
        unsigned short v = B[(size_t)gnc * ldb + gkc];
        dst[j] = (gn < N && gk < K) ? v : (unsigned short)0;
      }
      return 0;
    }
    // B row-major [K,N]: manual transpose stage (AV GEMM only)
    int kr = tid >> 3;            // 0..31
    int cs = (tid & 7) * 8;       // 0..56
    if (bColsIn && (k0 + BK <= K)) {
      const uint4* p = (const uint4*)(B + (size_t)(k0 + kr) * ldb + colBase + cs);
      uint4 q = *p;
      const unsigned short* e = (const unsigned short*)&q;
#pragma unroll
      for (int j = 0; j < 8; ++j) Bs[buf][(cs + j) * LDSS + kr] = e[j];
    } else {
      int gk  = k0 + kr;
      int gkc = gk < K ? gk : K - 1;
#pragma unroll
      for (int j = 0; j < 8; ++j) {
        int gn  = colBase + cs + j;
        int gnc = gn < N ? gn : N - 1;
        unsigned short v = B[(size_t)gkc * ldb + gnc];
        Bs[buf][(cs + j) * LDSS + kr] = (gk < K && gn < N) ? v : (unsigned short)0;
      }
    }
    return 0;
  };

  v8f acc[4] = {};

  // ---------------- double-buffered pipeline ----------------
  int buf = 0;
  (void)stageA(0, 0);
  (void)stageB(0, 0);

  for (int i = 0; i < ntiles; ++i) {
    int nextA = 0, nextT = 0;
    if (i + 1 < ntiles) {
      nextA = stageA((i + 1) * BK, buf ^ 1);
      nextT = stageB((i + 1) * BK, buf ^ 1);
    }
    // Drain tile i: in-order completion => wait until only tile i+1's
    // operations remain outstanding.
    if (nextA == 2)      asm volatile("s_wait_asynccnt 0x2" ::: "memory");
    else if (nextA == 1) asm volatile("s_wait_asynccnt 0x1" ::: "memory");
    else                 asm volatile("s_wait_asynccnt 0x0" ::: "memory");
#if HYB_TDM
    if (wid == 0) {
      if (nextT) __builtin_amdgcn_s_wait_tensorcnt((short)1);
      else       __builtin_amdgcn_s_wait_tensorcnt((short)0);
    }
#endif
    __syncthreads();

    const int kb = (lane >> 4) << 3;           // 0 or 8
    v16bf af = frag_from_lds(&As[buf][(wid * 16 + (lane & 15)) * LDSS + kb]);
#pragma unroll
    for (int t = 0; t < 4; ++t) {
      v16bf bfr = frag_from_lds(&Bs[buf][(t * 16 + (lane & 15)) * LDSS + kb]);
      acc[t] = __builtin_amdgcn_wmma_f32_16x16x32_bf16(
          false, af, false, bfr, (short)0, acc[t], false, false);
    }
    __syncthreads();
    buf ^= 1;
  }

  // ---- epilogue (C/D layout: lane<16 -> M=v, lane>=16 -> M=v+8) ----
  const int nl = lane & 15;
  const int ro = (lane >> 4) * 8;
  if (aRowsIn && bColsIn) {
#pragma unroll
    for (int t = 0; t < 4; ++t) {
      int gn = colBase + t * 16 + nl;
      float bv = (bias != nullptr) ? bias[gn] : 0.f;
#pragma unroll
      for (int v = 0; v < 8; ++v) {
        int gm = rowBase + wid * 16 + ro + v;
        float xv = acc[t][v] + bv;
        if (act == 1) xv = xv / (1.f + expf(-xv));   // silu
        if (C)   C[(size_t)gm * ldc + gn] = xv;
        if (Cbf) Cbf[(size_t)gm * ldcb + gn] = f2bf(xv);
      }
    }
  } else {
#pragma unroll
    for (int t = 0; t < 4; ++t) {
      int gn = colBase + t * 16 + nl;
      float bv = (bias != nullptr && gn < N) ? bias[gn] : 0.f;
#pragma unroll
      for (int v = 0; v < 8; ++v) {
        int gm = rowBase + wid * 16 + ro + v;
        if (gm < M && gn < N) {
          float xv = acc[t][v] + bv;
          if (act == 1) xv = xv / (1.f + expf(-xv));
          if (C)   C[(size_t)gm * ldc + gn] = xv;
          if (Cbf) Cbf[(size_t)gm * ldcb + gn] = f2bf(xv);
        }
      }
    }
  }
}

// ---------------------------------------------------------------------------
// Elementwise / reduction kernels
// ---------------------------------------------------------------------------

__global__ void hyb_copy(const float* __restrict__ in, float* __restrict__ out, int n) {
  int i = blockIdx.x * blockDim.x + threadIdx.x;
  if (i < n) out[i] = in[i];
}

// Wt[n*K + k] = bf16(W[k*N + n])  (transpose-convert weights once per launch)
__global__ void hyb_w2bft(const float* __restrict__ W, unsigned short* __restrict__ Wt,
                          int K, int N) {
  int i = blockIdx.x * blockDim.x + threadIdx.x;
  if (i >= K * N) return;
  int k = i / N, n = i % N;
  Wt[(size_t)n * K + k] = f2bf(W[i]);
}

__global__ __launch_bounds__(256) void hyb_rms_bf(
    const float* __restrict__ x, const float* __restrict__ w,
    unsigned short* __restrict__ out, int D) {
  int tok = blockIdx.x;
  const float* xr = x + (size_t)tok * D;
  float s = 0.f;
  for (int i = threadIdx.x; i < D; i += 256) { float v = xr[i]; s += v * v; }
#pragma unroll
  for (int o = 16; o > 0; o >>= 1) s += __shfl_xor(s, o, 32);
  __shared__ float red[8];
  int wid = threadIdx.x >> 5, ln = threadIdx.x & 31;
  if (ln == 0) red[wid] = s;
  __syncthreads();
  if (threadIdx.x == 0) {
    float t = 0.f;
    for (int i = 0; i < 8; ++i) t += red[i];
    red[0] = rsqrtf(t / (float)D + 1e-6f);
  }
  __syncthreads();
  float rs = red[0];
  for (int i = threadIdx.x; i < D; i += 256)
    out[(size_t)tok * D + i] = f2bf(w[i] * xr[i] * rs);
}

// gate = sigmoid(h[:, :1024]) f32; val = h[:, 1024:2048] f32
__global__ void hyb_split_gate(const float* __restrict__ h,
                               float* __restrict__ gate, float* __restrict__ val,
                               int total) {
  int i = blockIdx.x * blockDim.x + threadIdx.x;
  if (i >= total) return;
  int row = i >> 10;
  int col = i & 1023;
  float g = h[(size_t)row * 2048 + col];
  gate[i] = 1.f / (1.f + expf(-g));
  val[i]  = h[(size_t)row * 2048 + 1024 + col];
}

// depthwise conv along L; bf16 output (optionally with silu)
__global__ void hyb_dwconv_bf(const float* __restrict__ x, const float* __restrict__ w,
                              const float* __restrict__ bias, unsigned short* __restrict__ out,
                              int Bb, int L, int C, int ksz, int padl, int do_silu) {
  int i = blockIdx.x * blockDim.x + threadIdx.x;
  int total = Bb * L * C;
  if (i >= total) return;
  int c = i % C;
  int t = (i / C) % L;
  int b = i / (C * L);
  float acc = bias[c];
  for (int j = 0; j < ksz; ++j) {
    int tt = t - padl + j;
    if (tt >= 0 && tt < L) acc += x[((size_t)(b * L + tt)) * C + c] * w[c * ksz + j];
  }
  if (do_silu) acc = acc / (1.f + expf(-acc));
  out[i] = f2bf(acc);
}

__global__ void hyb_mul_bf(const float* __restrict__ a, const float* __restrict__ b,
                           unsigned short* __restrict__ out, int n) {
  int i = blockIdx.x * blockDim.x + threadIdx.x;
  if (i < n) out[i] = f2bf(a[i] * b[i]);
}

__global__ void hyb_add_inplace(float* __restrict__ x, const float* __restrict__ y, int n) {
  int i = blockIdx.x * blockDim.x + threadIdx.x;
  if (i < n) x[i] += y[i];
}

// probs(bf16) = softmax(s * scale) rowwise; s is scratch f32 (clobbered)
__global__ __launch_bounds__(256) void hyb_softmax_bf(
    float* __restrict__ s, unsigned short* __restrict__ p, int cols, float scale) {
  int row = blockIdx.x;
  float* r = s + (size_t)row * cols;
  unsigned short* pr = p + (size_t)row * cols;
  float mx = -3.4e38f;
  for (int i = threadIdx.x; i < cols; i += 256) mx = fmaxf(mx, r[i] * scale);
#pragma unroll
  for (int o = 16; o > 0; o >>= 1) mx = fmaxf(mx, __shfl_xor(mx, o, 32));
  __shared__ float red[8];
  __shared__ float bval;
  int wid = threadIdx.x >> 5, ln = threadIdx.x & 31;
  if (ln == 0) red[wid] = mx;
  __syncthreads();
  if (threadIdx.x == 0) {
    float m = red[0];
    for (int i = 1; i < 8; ++i) m = fmaxf(m, red[i]);
    bval = m;
  }
  __syncthreads();
  float m = bval;
  float sum = 0.f;
  for (int i = threadIdx.x; i < cols; i += 256) {
    float e = expf(r[i] * scale - m);
    r[i] = e;
    sum += e;
  }
#pragma unroll
  for (int o = 16; o > 0; o >>= 1) sum += __shfl_xor(sum, o, 32);
  if (ln == 0) red[wid] = sum;
  __syncthreads();
  if (threadIdx.x == 0) {
    float t = 0.f;
    for (int i = 0; i < 8; ++i) t += red[i];
    bval = 1.f / t;
  }
  __syncthreads();
  float inv = bval;
  for (int i = threadIdx.x; i < cols; i += 256) pr[i] = f2bf(r[i] * inv);
}

__global__ void hyb_top2(const float* __restrict__ logits, float* __restrict__ wdense,
                         int Ntok) {
  int n = blockIdx.x * blockDim.x + threadIdx.x;
  if (n >= Ntok) return;
  float p[8];
  float mx = -3.4e38f;
  for (int e = 0; e < 8; ++e) { p[e] = logits[n * 8 + e]; mx = fmaxf(mx, p[e]); }
  float sum = 0.f;
  for (int e = 0; e < 8; ++e) { p[e] = expf(p[e] - mx); sum += p[e]; }
  float inv = 1.f / sum;
  for (int e = 0; e < 8; ++e) p[e] *= inv;
  int i1 = 0;
  for (int e = 1; e < 8; ++e) if (p[e] > p[i1]) i1 = e;
  int i2 = (i1 == 0) ? 1 : 0;
  for (int e = 0; e < 8; ++e) if (e != i1 && p[e] > p[i2]) i2 = e;
  float s2 = p[i1] + p[i2] + 1e-9f;
  for (int e = 0; e < 8; ++e)
    wdense[n * 8 + e] = (e == i1) ? p[i1] / s2 : (e == i2) ? p[i2] / s2 : 0.f;
}

__global__ void hyb_moe_acc(float* __restrict__ x, const float* __restrict__ ye,
                            const float* __restrict__ wdense, int e, int total, int D) {
  int i = blockIdx.x * blockDim.x + threadIdx.x;
  if (i >= total) return;
  int n = i / D;
  x[i] += wdense[n * 8 + e] * ye[i];
}

// ---------------------------------------------------------------------------
// Host orchestration
// ---------------------------------------------------------------------------

static inline void launch_gemm(hipStream_t st,
                               const unsigned short* A, int lda,
                               const unsigned short* B, int ldb, int transB,
                               const float* bias,
                               float* C, int ldc, unsigned short* Cbf, int ldcb,
                               int M, int N, int K, int act) {
  dim3 g((unsigned)((N + BN - 1) / BN), (unsigned)((M + BM - 1) / BM));
  hyb_gemm_bf16<<<g, 256, 0, st>>>(A, lda, B, ldb, transB, bias, C, ldc, Cbf, ldcb,
                                   M, N, K, act);
}

static inline int eb(int n) { return (n + 255) / 256; }

extern "C" void kernel_launch(void* const* d_in, const int* in_sizes, int n_in,
                              void* d_out, int out_size, void* d_ws, size_t ws_size,
                              hipStream_t stream) {
  (void)in_sizes; (void)n_in; (void)out_size; (void)ws_size;

  const int Bb = 2, L = 2048, D = 512, DI = 1024;
  const int NT = Bb * L;           // 4096 tokens
  const int H = 8, HD = 64;

  int idx = 0;
  const float* x_in      = (const float*)d_in[idx++]; // 0
  const float* m_norm_w  = (const float*)d_in[idx++]; // 1  [2,512]
  const float* m_in_w    = (const float*)d_in[idx++]; // 2  [2,512,2048]
  const float* m_in_b    = (const float*)d_in[idx++]; // 3  [2,2048]
  const float* m_dwa_w   = (const float*)d_in[idx++]; // 4  [2,1024,1,4]
  const float* m_dwa_b   = (const float*)d_in[idx++]; // 5  [2,1024]
  const float* m_A       = (const float*)d_in[idx++]; // 6  [2,1024,16]
  const float* m_Bm      = (const float*)d_in[idx++]; // 7  [2,16,1024]
  const float* m_sdw_w   = (const float*)d_in[idx++]; // 8  [2,1024,1,9]
  const float* m_sdw_b   = (const float*)d_in[idx++]; // 9  [2,1024]
  const float* m_spw_w   = (const float*)d_in[idx++]; // 10 [2,1024,1024]
  const float* m_spw_b   = (const float*)d_in[idx++]; // 11 [2,1024]
  const float* m_ha_w    = (const float*)d_in[idx++]; // 12 [2,1024,512]
  const float* m_ha_b    = (const float*)d_in[idx++]; // 13 [2,512]
  const float* m_hb_w    = (const float*)d_in[idx++]; // 14 [2,1024,512]
  const float* m_hb_b    = (const float*)d_in[idx++]; // 15 [2,512]
  const float* m_fuse_w  = (const float*)d_in[idx++]; // 16 [2,1024,1024]
  const float* m_fuse_b  = (const float*)d_in[idx++]; // 17 [2,1024]
  const float* m_out_w   = (const float*)d_in[idx++]; // 18 [2,1024,512]
  const float* m_out_b   = (const float*)d_in[idx++]; // 19 [2,512]
  const float* a_n1_w    = (const float*)d_in[idx++]; // 20 [512]
  const float* a_wqkv    = (const float*)d_in[idx++]; // 21 [512,1536]
  const float* a_bqkv    = (const float*)d_in[idx++]; // 22 [1536]
  const float* a_wo      = (const float*)d_in[idx++]; // 23 [512,512]
  const float* a_bo      = (const float*)d_in[idx++]; // 24 [512]
  const float* a_proj_w  = (const float*)d_in[idx++]; // 25 [512,512]
  const float* a_proj_b  = (const float*)d_in[idx++]; // 26 [512]
  const float* a_n2_w    = (const float*)d_in[idx++]; // 27 [512]
  const float* e_rout_w  = (const float*)d_in[idx++]; // 28 [512,8]
  const float* e_fc1_w   = (const float*)d_in[idx++]; // 29 [8,512,2048]
  const float* e_fc1_b   = (const float*)d_in[idx++]; // 30 [8,2048]
  const float* e_fc2_w   = (const float*)d_in[idx++]; // 31 [8,2048,512]
  const float* e_fc2_b   = (const float*)d_in[idx++]; // 32 [8,512]

  float* x = (float*)d_out;           // running activations [NT, D]

  // ---------------- workspace carving (byte cursor) ----------------
  char* base = (char*)d_ws;
  size_t o = 0;
  auto take = [&](size_t bytes) -> char* {
    char* p = base + o;
    o += (bytes + 255) & ~(size_t)255;
    return p;
  };

  // persistent: transposed bf16 weights [N,K]
  unsigned short* wt_in[2];   unsigned short* wt_Am[2];  unsigned short* wt_Bm[2];
  unsigned short* wt_spw[2];  unsigned short* wt_ha[2];  unsigned short* wt_hb[2];
  unsigned short* wt_fuse[2]; unsigned short* wt_out[2];
  for (int l = 0; l < 2; ++l) {
    wt_in[l]   = (unsigned short*)take((size_t)2048 * 512  * 2);
    wt_Am[l]   = (unsigned short*)take((size_t)16   * 1024 * 2);
    wt_Bm[l]   = (unsigned short*)take((size_t)1024 * 16   * 2);
    wt_spw[l]  = (unsigned short*)take((size_t)1024 * 1024 * 2);
    wt_ha[l]   = (unsigned short*)take((size_t)512  * 1024 * 2);
    wt_hb[l]   = (unsigned short*)take((size_t)512  * 1024 * 2);
    wt_fuse[l] = (unsigned short*)take((size_t)1024 * 1024 * 2);
    wt_out[l]  = (unsigned short*)take((size_t)512  * 1024 * 2);
  }
  unsigned short* wt_qkv  = (unsigned short*)take((size_t)1536 * 512 * 2);
  unsigned short* wt_wo   = (unsigned short*)take((size_t)512 * 512 * 2);
  unsigned short* wt_proj = (unsigned short*)take((size_t)512 * 512 * 2);
  unsigned short* wt_rout = (unsigned short*)take((size_t)8 * 512 * 2);
  unsigned short* wt_fc1  = (unsigned short*)take((size_t)8 * 2048 * 512 * 2);
  unsigned short* wt_fc2  = (unsigned short*)take((size_t)8 * 512 * 2048 * 2);

  unsigned short* xnbf = (unsigned short*)take((size_t)NT * 512 * 2);

  size_t R = o;   // phase regions overlay from here
  // --- mamba region ---
  float*          hin     = (float*)take((size_t)NT * 2048 * 4);
  float*          gate    = (float*)take((size_t)NT * 1024 * 4);
  float*          val     = (float*)take((size_t)NT * 1024 * 4);
  unsigned short* aconvbf = (unsigned short*)take((size_t)NT * 1024 * 2);
  unsigned short* t16bf   = (unsigned short*)take((size_t)NT * 16 * 2);
  unsigned short* alrbf   = (unsigned short*)take((size_t)NT * 1024 * 2);
  unsigned short* silubsbf= (unsigned short*)take((size_t)NT * 1024 * 2);
  unsigned short* bsactbf = (unsigned short*)take((size_t)NT * 1024 * 2);
  unsigned short* catbf   = (unsigned short*)take((size_t)NT * 1024 * 2);
  float*          fuse    = (float*)take((size_t)NT * 1024 * 4);
  unsigned short* gvalbf  = (unsigned short*)take((size_t)NT * 1024 * 2);
  float*          ybuf    = (float*)take((size_t)NT * 512 * 4);
  // --- attention region (overlay) ---
  o = R;
  unsigned short* qkvbf   = (unsigned short*)take((size_t)NT * 1536 * 2);
  float*          scores  = (float*)take((size_t)L * L * 4);
  unsigned short* probsbf = (unsigned short*)take((size_t)L * L * 2);
  unsigned short* obufbf  = (unsigned short*)take((size_t)NT * 512 * 2);
  unsigned short* t2bf    = (unsigned short*)take((size_t)NT * 512 * 2);
  float*          o2      = (float*)take((size_t)NT * 512 * 4);
  // --- moe region (overlay) ---
  o = R;
  float*          logits  = (float*)take((size_t)NT * 8 * 4);
  float*          wdense  = (float*)take((size_t)NT * 8 * 4);
  unsigned short* moehbf  = (unsigned short*)take((size_t)NT * 2048 * 2);
  float*          ye      = (float*)take((size_t)NT * 512 * 4);

  // ---------------- weight transpose-convert (f32 -> bf16 [N,K]) ----------------
  for (int l = 0; l < 2; ++l) {
    hyb_w2bft<<<eb(512 * 2048), 256, 0, stream>>>(m_in_w  + (size_t)l * 512 * 2048, wt_in[l],   512, 2048);
    hyb_w2bft<<<eb(1024 * 16),  256, 0, stream>>>(m_A     + (size_t)l * 1024 * 16,  wt_Am[l],  1024, 16);
    hyb_w2bft<<<eb(16 * 1024),  256, 0, stream>>>(m_Bm    + (size_t)l * 16 * 1024,  wt_Bm[l],    16, 1024);
    hyb_w2bft<<<eb(1024 * 1024),256, 0, stream>>>(m_spw_w + (size_t)l * 1024 * 1024,wt_spw[l], 1024, 1024);
    hyb_w2bft<<<eb(1024 * 512), 256, 0, stream>>>(m_ha_w  + (size_t)l * 1024 * 512, wt_ha[l],  1024, 512);
    hyb_w2bft<<<eb(1024 * 512), 256, 0, stream>>>(m_hb_w  + (size_t)l * 1024 * 512, wt_hb[l],  1024, 512);
    hyb_w2bft<<<eb(1024 * 1024),256, 0, stream>>>(m_fuse_w+ (size_t)l * 1024 * 1024,wt_fuse[l],1024, 1024);
    hyb_w2bft<<<eb(1024 * 512), 256, 0, stream>>>(m_out_w + (size_t)l * 1024 * 512, wt_out[l], 1024, 512);
  }
  hyb_w2bft<<<eb(512 * 1536), 256, 0, stream>>>(a_wqkv,   wt_qkv,  512, 1536);
  hyb_w2bft<<<eb(512 * 512),  256, 0, stream>>>(a_wo,     wt_wo,   512, 512);
  hyb_w2bft<<<eb(512 * 512),  256, 0, stream>>>(a_proj_w, wt_proj, 512, 512);
  hyb_w2bft<<<eb(512 * 8),    256, 0, stream>>>(e_rout_w, wt_rout, 512, 8);
  for (int e = 0; e < 8; ++e) {
    hyb_w2bft<<<eb(512 * 2048), 256, 0, stream>>>(e_fc1_w + (size_t)e * 512 * 2048,
                                                  wt_fc1 + (size_t)e * 2048 * 512, 512, 2048);
    hyb_w2bft<<<eb(2048 * 512), 256, 0, stream>>>(e_fc2_w + (size_t)e * 2048 * 512,
                                                  wt_fc2 + (size_t)e * 512 * 2048, 2048, 512);
  }

  // x = input
  hyb_copy<<<eb(NT * D), 256, 0, stream>>>(x_in, x, NT * D);

  // -------------------- Mamba blocks --------------------
  for (int l = 0; l < 2; ++l) {
    const float* nw   = m_norm_w + (size_t)l * 512;
    const float* inb  = m_in_b   + (size_t)l * 2048;
    const float* dwaw = m_dwa_w  + (size_t)l * 1024 * 4;
    const float* dwab = m_dwa_b  + (size_t)l * 1024;
    const float* sdww = m_sdw_w  + (size_t)l * 1024 * 9;
    const float* sdwb = m_sdw_b  + (size_t)l * 1024;
    const float* spwb = m_spw_b  + (size_t)l * 1024;
    const float* hab  = m_ha_b   + (size_t)l * 512;
    const float* hbb  = m_hb_b   + (size_t)l * 512;
    const float* fb   = m_fuse_b + (size_t)l * 1024;
    const float* ob   = m_out_b  + (size_t)l * 512;

    hyb_rms_bf<<<NT, 256, 0, stream>>>(x, nw, xnbf, D);
    launch_gemm(stream, xnbf, 512, wt_in[l], 512, 1, inb, hin, 2048, nullptr, 0,
                NT, 2048, 512, 0);
    hyb_split_gate<<<eb(NT * DI), 256, 0, stream>>>(hin, gate, val, NT * DI);
    hyb_dwconv_bf<<<eb(NT * DI), 256, 0, stream>>>(val, dwaw, dwab, aconvbf, Bb, L, DI, 4, 1, 0);
    launch_gemm(stream, aconvbf, 1024, wt_Am[l], 1024, 1, nullptr, nullptr, 0, t16bf, 16,
                NT, 16, 1024, 0);
    launch_gemm(stream, t16bf, 16, wt_Bm[l], 16, 1, nullptr, nullptr, 0, alrbf, 1024,
                NT, 1024, 16, 0);
    hyb_dwconv_bf<<<eb(NT * DI), 256, 0, stream>>>(val, sdww, sdwb, silubsbf, Bb, L, DI, 9, 4, 1);
    launch_gemm(stream, silubsbf, 1024, wt_spw[l], 1024, 1, spwb, nullptr, 0, bsactbf, 1024,
                NT, 1024, 1024, 0);
    launch_gemm(stream, alrbf, 1024, wt_ha[l], 1024, 1, hab, nullptr, 0, catbf, 1024,
                NT, 512, 1024, 0);
    launch_gemm(stream, bsactbf, 1024, wt_hb[l], 1024, 1, hbb, nullptr, 0, catbf + 512, 1024,
                NT, 512, 1024, 0);
    launch_gemm(stream, catbf, 1024, wt_fuse[l], 1024, 1, fb, fuse, 1024, nullptr, 0,
                NT, 1024, 1024, 0);
    hyb_mul_bf<<<eb(NT * DI), 256, 0, stream>>>(gate, fuse, gvalbf, NT * DI);
    launch_gemm(stream, gvalbf, 1024, wt_out[l], 1024, 1, ob, ybuf, 512, nullptr, 0,
                NT, 512, 1024, 0);
    hyb_add_inplace<<<eb(NT * D), 256, 0, stream>>>(x, ybuf, NT * D);
  }

  // -------------------- Attention block --------------------
  hyb_rms_bf<<<NT, 256, 0, stream>>>(x, a_n1_w, xnbf, D);
  launch_gemm(stream, xnbf, 512, wt_qkv, 512, 1, a_bqkv, nullptr, 0, qkvbf, 1536,
              NT, 1536, 512, 0);

  for (int b = 0; b < Bb; ++b) {
    for (int h = 0; h < H; ++h) {
      const unsigned short* Q  = qkvbf + (size_t)b * L * 1536 + h * HD;
      const unsigned short* Kp = Q + 512;
      const unsigned short* V  = Q + 1024;
      launch_gemm(stream, Q, 1536, Kp, 1536, 1, nullptr, scores, L, nullptr, 0,
                  L, L, HD, 0);
      hyb_softmax_bf<<<L, 256, 0, stream>>>(scores, probsbf, L, 0.125f); // 1/sqrt(64)
      launch_gemm(stream, probsbf, L, V, 1536, 0, nullptr, nullptr, 0,
                  obufbf + (size_t)b * L * 512 + h * HD, 512, L, HD, L, 0);
    }
  }

  launch_gemm(stream, obufbf, 512, wt_wo, 512, 1, a_bo, nullptr, 0, t2bf, 512,
              NT, 512, 512, 0);
  launch_gemm(stream, t2bf, 512, wt_proj, 512, 1, a_proj_b, o2, 512, nullptr, 0,
              NT, 512, 512, 0);
  hyb_add_inplace<<<eb(NT * D), 256, 0, stream>>>(x, o2, NT * D);

  // -------------------- MoE --------------------
  hyb_rms_bf<<<NT, 256, 0, stream>>>(x, a_n2_w, xnbf, D);
  launch_gemm(stream, xnbf, 512, wt_rout, 512, 1, nullptr, logits, 8, nullptr, 0,
              NT, 8, 512, 0);
  hyb_top2<<<eb(NT), 256, 0, stream>>>(logits, wdense, NT);

  for (int e = 0; e < 8; ++e) {
    launch_gemm(stream, xnbf, 512, wt_fc1 + (size_t)e * 2048 * 512, 512, 1,
                e_fc1_b + (size_t)e * 2048, nullptr, 0, moehbf, 2048,
                NT, 2048, 512, 1);                              // silu
    launch_gemm(stream, moehbf, 2048, wt_fc2 + (size_t)e * 512 * 2048, 2048, 1,
                e_fc2_b + (size_t)e * 512, ye, 512, nullptr, 0,
                NT, 512, 2048, 0);
    hyb_moe_acc<<<eb(NT * D), 256, 0, stream>>>(x, ye, wdense, e, NT * D, D);
  }
}
